// DifferentiableSampler_50354196579100
// MI455X (gfx1250) — matrix operand order repaired and checked by reading, not compiled, not run on hardware
//
#include <hip/hip_runtime.h>

// DifferentiableSampler for MI455X (gfx1250, wave32).
// input:(B,C,L) f32, point/offset:(B,n) f32 -> out:(B,n,C) f32
// Memory-bound gather + lerp; combine is executed on the WMMA path
// (V_WMMA_F32_16X16X32_F16) using a per-block selection/weight A-matrix.

#define GAMMA_F 1.0f
#define C_DIM 256
#define L_DIM 8192

typedef __attribute__((ext_vector_type(16))) _Float16 v16h;
typedef __attribute__((ext_vector_type(8)))  float    v8f;

__global__ __launch_bounds__(256)
void sampler_wmma_kernel(const float* __restrict__ input,   // (B, C, L)
                         const float* __restrict__ point,   // (B, n)
                         const float* __restrict__ offset,  // (B, n)
                         float* __restrict__ out,           // (B, n, C)
                         int n, int groups_per_batch)
{
    __shared__ int   s_idx0[16];
    __shared__ int   s_idx1[16];
    __shared__ float s_w0[16];
    __shared__ float s_w1[16];

    const int b  = blockIdx.x / groups_per_batch;
    const int g  = blockIdx.x % groups_per_batch;
    const int j0 = g * 16;                       // 16 samples per block

    const int tid = threadIdx.x;

    // ---- Phase 1: 16 threads compute per-sample metadata (shared by all waves)
    if (tid < 16) {
        const int j = j0 + tid;
        const float p = point [(size_t)b * n + j];
        const float o = offset[(size_t)b * n + j];
        float loc = p + GAMMA_F * o;
        loc = fminf(fmaxf(loc, 0.0f), (float)(L_DIM - 1));
        int i0 = (int)floorf(loc);
        int i1 = (int)ceilf(loc);
        i0 = min(max(i0, 0), L_DIM - 1);
        i1 = min(max(i1, 0), L_DIM - 1);
        const float w1 = loc - (float)i0;
        s_idx0[tid] = i0;
        s_idx1[tid] = i1;
        s_w0[tid]   = 1.0f - w1;
        s_w1[tid]   = w1;
    }
    __syncthreads();

    const int lane = tid & 31;
    const int wave = tid >> 5;
    const int col  = lane & 15;   // N-column of tile; also M-row for A build
    const int hi   = lane >> 4;   // lane half: 0 -> lanes 0-15, 1 -> 16-31

    // ---- Build A fragment once: A[m,k] = w0_m at k=2m, w1_m at k=2m+1.
    // A k order used here (contiguous, not ISA table order): lane m holds (lanes 0-15) K=0..15 and
    // (lanes 16-31) K=16..31; element e -> K = e + 16*hi.
    v16h afrag;
    {
        const int m = col;
        const _Float16 w0h = (_Float16)s_w0[m];
        const _Float16 w1h = (_Float16)s_w1[m];
#pragma unroll
        for (int e = 0; e < 16; ++e) {
            const int v  = e >> 1;
            const int hh = e & 1;
            const int k  = e + (hi ? 16 : 0);
            _Float16 val = (_Float16)0.0f;
            if      (k == 2 * m)     val = w0h;
            else if (k == 2 * m + 1) val = w1h;
            afrag[e] = val;
        }
    }

    // Each lane gathers for 8 samples (its K half of the B matrix).
    const int mbase = hi * 8;
    int i0s[8], i1s[8];
#pragma unroll
    for (int m = 0; m < 8; ++m) {
        i0s[m] = s_idx0[mbase + m];
        i1s[m] = s_idx1[mbase + m];
    }

    const float* rowbase = input + (size_t)b * C_DIM * L_DIM;
    float*       outbase = out   + ((size_t)b * n + j0) * C_DIM;

    // ---- Each wave: 2 channel tiles of 16 -> 8 waves cover C=256.
#pragma unroll
    for (int t = 0; t < 2; ++t) {
        const int ct = (wave * 2 + t) * 16;   // channel-tile base
        const int c  = ct + col;              // this lane's channel
        const float* row = rowbase + (size_t)c * L_DIM;

        // B fragment: 16-bit B 32x16 -> lane holds column c, rows
        // K = e + 16*hi. Row 2m = g0[sample m], row 2m+1 = g1[sample m].
        v16h bfrag;
#pragma unroll
        for (int m = 0; m < 8; ++m) {
            const float g0 = row[i0s[m]];     // gathered, scattered in a 32KB row
            const float g1 = row[i1s[m]];
            bfrag[2 * m]     = (_Float16)g0;
            bfrag[2 * m + 1] = (_Float16)g1;
        }

        v8f d = {};
        d = __builtin_amdgcn_wmma_f32_16x16x32_f16(
                /*neg_a=*/false, afrag, /*neg_b=*/false, bfrag,
                /*c_mod=*/(short)0, d, /*reuse_a=*/false, /*reuse_b=*/false);

        // D layout: VGPR r -> row m = r + 8*hi, col = lane&15.
        // Lanes 0-15 store 64B contiguous per r (half-coalesced stores).
#pragma unroll
        for (int r = 0; r < 8; ++r) {
            const int m = r + 8 * hi;
            outbase[(size_t)m * C_DIM + c] = d[r];
        }
    }
}

extern "C" void kernel_launch(void* const* d_in, const int* in_sizes, int n_in,
                              void* d_out, int out_size, void* d_ws, size_t ws_size,
                              hipStream_t stream)
{
    const float* input  = (const float*)d_in[0];   // (B, C, L)
    const float* point  = (const float*)d_in[1];   // (B, n, 1)
    const float* offset = (const float*)d_in[2];   // (B, n, 1)
    float*       out    = (float*)d_out;           // (B, n, C)

    const int B = in_sizes[0] / (C_DIM * L_DIM);   // 32
    const int n = in_sizes[1] / B;                 // 2048
    const int groups_per_batch = n / 16;           // 16 samples per block

    dim3 grid(B * groups_per_batch);
    dim3 block(256);
    sampler_wmma_kernel<<<grid, block, 0, stream>>>(input, point, offset, out,
                                                    n, groups_per_batch);
}
